// Router_36876589204273
// MI455X (gfx1250) — matrix operand order repaired and checked by reading, compiled, not run on hardware
//
#include <hip/hip_runtime.h>
#include <stdint.h>

// ---------------------------------------------------------------------------
// MoE router for MI455X (gfx1250, wave32).
// logits = x @ gate_w^T via split-precision bf16 WMMA (AhiBhi + AhiBlo + AloBhi),
// K split 2-way across waves for occupancy, fused softmax + top-8 per token.
// ---------------------------------------------------------------------------

#define DDIM 2048
#define EEXP 64
#define NSEL 8
#define KCHUNKS (DDIM / 32)   // 64 chunks of K=32
#define KSPLIT 2              // K-halves per token tile (occupancy: 2048 waves)
#define PFDIST 8              // A-prefetch distance in chunks (~1 KB/row ahead)

typedef __attribute__((ext_vector_type(16))) __bf16 v16bf;
typedef __attribute__((ext_vector_type(8)))  float  v8f;
typedef __attribute__((ext_vector_type(4)))  float  f4;

union V16U {            // one WMMA 16-bit A/B fragment = 8 VGPRs = 32 bytes
    v16bf    v;
    uint32_t u[8];
    f4       f[2];
};

static __device__ inline uint16_t bf16_bits(__bf16 h) {
    union { __bf16 h; uint16_t u; } c; c.h = h; return c.u;
}

// ---------------------------------------------------------------------------
// Prep: split gate_w (fp32, row-major [E, D]) into hi/lo bf16, stored directly
// in WMMA B-fragment order (ISA 7.12.2, 16-bit B 32x16):
//   lane 0-15  -> N = lane,    K = 0..7,16..23 of the chunk (VGPR j<4: K=2j,2j+1; j>=4: K=2j+8,2j+9)
//   lane 16-31 -> N = lane-16, K = 8..15,24..31 of the chunk
// Layout: u32[ ((chunk*4 + tile)*32 + lane)*8 + j ]
// ---------------------------------------------------------------------------
__global__ void router_prep_bfrag(const float* __restrict__ gw,
                                  uint32_t* __restrict__ bhi,
                                  uint32_t* __restrict__ blo) {
    int idx = blockIdx.x * blockDim.x + threadIdx.x;   // one per (chunk, tile, lane)
    if (idx >= KCHUNKS * 4 * 32) return;
    int l = idx & 31;
    int t = (idx >> 5) & 3;
    int c = idx >> 7;
    int e  = t * 16 + (l & 15);
    int k0 = c * 32 + ((l & 16) ? 8 : 0);
    const float* src = gw + (size_t)e * DDIM + k0;
    uint32_t* dh = bhi + (size_t)idx * 8;
    uint32_t* dl = blo + (size_t)idx * 8;
#pragma unroll
    for (int j = 0; j < 8; ++j) {
        float f0 = src[2 * j + (j & 4) * 2], f1 = src[2 * j + (j & 4) * 2 + 1];
        uint32_t u0 = __float_as_uint(f0), u1 = __float_as_uint(f1);
        uint32_t h0 = u0 & 0xFFFF0000u,   h1 = u1 & 0xFFFF0000u;
        dh[j] = (h0 >> 16) | h1;                        // packed (even K low half)
        __bf16 l0 = (__bf16)(f0 - __uint_as_float(h0)); // exact residual -> bf16
        __bf16 l1 = (__bf16)(f1 - __uint_as_float(h1));
        dl[j] = (uint32_t)bf16_bits(l0) | ((uint32_t)bf16_bits(l1) << 16);
    }
}

// ---------------------------------------------------------------------------
// One K=32 chunk: load A (two 32B runs/lane), hi/lo split, 4 expert tiles x
// 3 split-precision WMMAs.  DO_PF=true issues an unconditional A prefetch
// PFDIST chunks ahead (caller guarantees it stays in range).
// ---------------------------------------------------------------------------
template <bool DO_PF>
static __device__ __forceinline__ void chunk_step(int c,
                                                  const float* __restrict__ xrow,
                                                  const uint32_t* __restrict__ bhi,
                                                  const uint32_t* __restrict__ blo,
                                                  int lane, v8f (&acc)[4]) {
    if (DO_PF)
        __builtin_prefetch(xrow + (c + PFDIST) * 32, 0, 3);

    union { f4 v[4]; float s[16]; } A;
    A.v[0] = *(const f4*)(xrow + c * 32 + 0);
    A.v[1] = *(const f4*)(xrow + c * 32 + 4);
    A.v[2] = *(const f4*)(xrow + c * 32 + 16);
    A.v[3] = *(const f4*)(xrow + c * 32 + 20);

    // split A into hi (top 16 bits, exact) + lo (residual, bf16-rounded)
    V16U ah, al;
#pragma unroll
    for (int i = 0; i < 8; ++i) {
        uint32_t u0 = __float_as_uint(A.s[2 * i]);
        uint32_t u1 = __float_as_uint(A.s[2 * i + 1]);
        uint32_t h0 = u0 & 0xFFFF0000u, h1 = u1 & 0xFFFF0000u;
        ah.u[i] = (h0 >> 16) | h1;
        al.v[2 * i]     = (__bf16)(A.s[2 * i]     - __uint_as_float(h0));
        al.v[2 * i + 1] = (__bf16)(A.s[2 * i + 1] - __uint_as_float(h1));
    }

#pragma unroll
    for (int t = 0; t < 4; ++t) {
        const size_t off = ((size_t)(c * 4 + t) * 32 + lane) * 8;
        V16U bh, bl;
        bh.f[0] = ((const f4*)(bhi + off))[0];
        bh.f[1] = ((const f4*)(bhi + off))[1];
        bl.f[0] = ((const f4*)(blo + off))[0];
        bl.f[1] = ((const f4*)(blo + off))[1];

        acc[t] = __builtin_amdgcn_wmma_f32_16x16x32_bf16(
            false, ah.v, false, bh.v, (short)0, acc[t], false, false);
        acc[t] = __builtin_amdgcn_wmma_f32_16x16x32_bf16(
            false, ah.v, false, bl.v, (short)0, acc[t], false, false);
        acc[t] = __builtin_amdgcn_wmma_f32_16x16x32_bf16(
            false, al.v, false, bh.v, (short)0, acc[t], false, false);
    }
}

// ---------------------------------------------------------------------------
// Main fused kernel: 128 threads = 4 waves = 2 token tiles x 2 K-halves.
// Wave (tile,khalf) accumulates tokens [blk*32 + tile*16, +16) over K-half
// khalf. Partial C tiles land in separate LDS buffers, combined in phase 2.
// ---------------------------------------------------------------------------
__global__ __launch_bounds__(128)
void router_gemm_softmax_topk(const float* __restrict__ x,
                              const uint32_t* __restrict__ bhi,
                              const uint32_t* __restrict__ blo,
                              float* __restrict__ topp,
                              float* __restrict__ topi,
                              float* __restrict__ allp) {
    // [KSPLIT][32 tokens][65] : stride-65 rows -> conflict-free column reads
    __shared__ float lds[KSPLIT * 32 * 65];

    const int tid    = threadIdx.x;
    const int lane   = tid & 31;
    const int wave   = tid >> 5;
    const int tile   = wave >> 1;          // which 16-token tile (0..1)
    const int khalf  = wave & 1;           // which K half (0..1)
    const int wgBase = blockIdx.x * 32;

    // A fragment mapping (ISA 7.12.2, 16-bit A 16x32): lane%16 = row M,
    // kbase = lane<16 ? 0 : 8; K runs [kb, kb+8) (VGPR0-3), [kb+16, kb+24) (VGPR4-7).
    const int tok = wgBase + tile * 16 + (lane & 15);
    const int kb  = (lane & 16) ? 8 : 0;
    const float* xrow = x + (size_t)tok * DDIM + kb;

    const int c0 = khalf * (KCHUNKS / KSPLIT);
    const int c1 = c0 + (KCHUNKS / KSPLIT);

    // Prime the A prefetch pipeline (global_prefetch_b8: pulls the 128B line
    // of each token row for the chunk; no VGPR result, no counter).
#pragma unroll
    for (int p = 0; p < PFDIST; ++p)
        __builtin_prefetch(xrow + (c0 + p) * 32, 0, 3);

    v8f acc[4] = {};   // 4 expert tiles

    // Steady state: unconditional prefetch (no per-iteration branch), then
    // a PFDIST-chunk tail without prefetch (stays in-bounds).
    int c = c0;
    for (; c < c1 - PFDIST; ++c)
        chunk_step<true>(c, xrow, bhi, blo, lane, acc);
    for (; c < c1; ++c)
        chunk_step<false>(c, xrow, bhi, blo, lane, acc);

    // ---- scatter partial logits to LDS.  C tile layout (ISA 7.12.2):
    // VGPR r, lane l holds element (M = r + 8*(l>=16), N = l%16).
    {
        const int mhalf = (lane >> 4) * 8;
        const int ncol  = lane & 15;
        float* buf = &lds[khalf * (32 * 65)];
#pragma unroll
        for (int t = 0; t < 4; ++t) {
#pragma unroll
            for (int r = 0; r < 8; ++r) {
                buf[(tile * 16 + mhalf + r) * 65 + (t * 16 + ncol)] = acc[t][r];
            }
        }
    }
    __syncthreads();

    // ---- combine K-halves + softmax + top-8: one thread per token
    if (tid < 32) {
        float* row        = &lds[tid * 65];
        const float* rowb = &lds[32 * 65 + tid * 65];
        const int token   = wgBase + tid;

#pragma unroll
        for (int e = 0; e < EEXP; ++e) row[e] += rowb[e];

        float mx = row[0];
#pragma unroll
        for (int e = 1; e < EEXP; ++e) mx = fmaxf(mx, row[e]);
        float sum = 0.0f;
#pragma unroll
        for (int e = 0; e < EEXP; ++e) { float p = __expf(row[e] - mx); row[e] = p; sum += p; }
        const float inv = 1.0f / sum;
#pragma unroll
        for (int e = 0; e < EEXP; ++e) row[e] *= inv;

        // all_probs, vectorized 16B stores
#pragma unroll
        for (int e = 0; e < EEXP; e += 4) {
            f4 v = { row[e], row[e + 1], row[e + 2], row[e + 3] };
            *(f4*)(allp + (size_t)token * EEXP + e) = v;
        }

        // top-8 selection (descending, lowest index on ties — matches lax.top_k)
        uint64_t used = 0;
        float tv[NSEL]; int ti[NSEL];
        float tsum = 0.0f;
#pragma unroll 1
        for (int k = 0; k < NSEL; ++k) {
            float best = -1.0f; int bi = 0;
            for (int e = 0; e < EEXP; ++e) {
                float p = row[e];
                if (!((used >> e) & 1ull) && p > best) { best = p; bi = e; }
            }
            used |= 1ull << bi;
            tv[k] = best; ti[k] = bi; tsum += best;
        }
        const float scale = 1.0f / (tsum + 1e-9f);
#pragma unroll
        for (int k = 0; k < NSEL; ++k) {
            topp[(size_t)token * NSEL + k] = tv[k] * scale;
            topi[(size_t)token * NSEL + k] = (float)ti[k];
        }
    }
}

// ---------------------------------------------------------------------------
extern "C" void kernel_launch(void* const* d_in, const int* in_sizes, int n_in,
                              void* d_out, int out_size, void* d_ws, size_t ws_size,
                              hipStream_t stream) {
    (void)n_in; (void)out_size; (void)ws_size;
    const float* x  = (const float*)d_in[0];
    const float* gw = (const float*)d_in[1];
    const int BS = in_sizes[0] / DDIM;                 // 16384 tokens

    float* topp = (float*)d_out;
    float* topi = topp + (size_t)BS * NSEL;
    float* allp = topi + (size_t)BS * NSEL;

    // ws: pre-swizzled B fragments, hi + lo (2 x 256 KB), L2-resident
    uint32_t* bhi = (uint32_t*)d_ws;
    uint32_t* blo = bhi + (size_t)KCHUNKS * 4 * 32 * 8;

    const int prepN = KCHUNKS * 4 * 32;                // 8192
    router_prep_bfrag<<<prepN / 256, 256, 0, stream>>>(gw, bhi, blo);

    // 512 WGs x 4 waves = 2048 waves (~2 per SIMD on a ~256-WGP part)
    router_gemm_softmax_topk<<<BS / 32, 128, 0, stream>>>(x, bhi, blo, topp, topi, allp);
}
